// HRNR_6751688589918
// MI455X (gfx1250) — matrix operand order repaired and checked by reading, not compiled, not run on hardware
//
#include <hip/hip_runtime.h>
#include <hip/hip_bf16.h>

// ---------------------------------------------------------------------------
// HRNR pipeline for MI455X (gfx1250, wave32).
// Dense GEMMs -> v_wmma_f32_16x16x32_bf16 (bf16 operands, f32 accum).
// Sparse GAT  -> wave-per-edge gather + f32 global atomics.
// ---------------------------------------------------------------------------

typedef __attribute__((ext_vector_type(16))) __bf16 v16bf;
typedef __attribute__((ext_vector_type(8)))  __bf16 v8bf;
typedef __attribute__((ext_vector_type(8)))  float  v8f;

static __device__ __forceinline__ unsigned short f2bf_bits(float f) {
  unsigned u = __builtin_bit_cast(unsigned, f);
  return (unsigned short)(u >> 16);   // truncate-to-bf16
}
// pack two floats -> bf16x2 dword (lo = first element, little-endian)
static __device__ __forceinline__ unsigned pack2bf(float a, float b) {
  unsigned ua = __builtin_bit_cast(unsigned, a);
  unsigned ub = __builtin_bit_cast(unsigned, b);
  return (ua >> 16) | (ub & 0xffff0000u);
}

// monotone float<->uint encoding for atomicMax over signed floats
static __device__ __forceinline__ unsigned enc_f32(float f) {
  int i = __float_as_int(f);
  return (i >= 0) ? ((unsigned)i | 0x80000000u) : ~(unsigned)i;
}
static __device__ __forceinline__ float dec_f32(unsigned k) {
  int i = (k & 0x80000000u) ? (int)(k & 0x7fffffffu) : (int)~k;
  return __int_as_float(i);
}

// ---------------------------------------------------------------------------
// WMMA GEMM: C[M x 256] = epilogue( op(A)[M x Kr] @ B[Kr x 256] )
//   AT=false: A is row-major [M x Kr], lda = row stride (must be even)
//   AT=true : source S is [Kr x M] row-major (lda = M stride), op(A)=S^T
//   epilogue: out = alpha*rowScale[m]*acc + beta*Cin[m,n] + bias[n]
// Block: 256 threads = 8 waves; block tile 64 rows x 256 cols (full output
// width so the big A matrices stream from HBM exactly once).
// ---------------------------------------------------------------------------
#define AS_STRIDE 40   // halves; 80B rows -> 16B-aligned fragments, low conflicts
#define BS_STRIDE 40

static __device__ __forceinline__ v16bf load_frag16(const unsigned short* p0,
                                                    const unsigned short* p1) {
  v8bf lo = *(const v8bf*)p0;
  v8bf hi = *(const v8bf*)p1;
  return __builtin_shufflevector(lo, hi, 0,1,2,3,4,5,6,7,8,9,10,11,12,13,14,15);
}

template <bool AT>
__global__ __launch_bounds__(256)
void gemm256(const float* __restrict__ A, int lda,
             const float* __restrict__ B,
             float* __restrict__ C,
             int Mdim, int Kr,
             float alpha, const float* __restrict__ rowScale,
             float beta,  const float* __restrict__ Cin,
             const float* __restrict__ bias)
{
  __shared__ __align__(16) unsigned short As[64  * AS_STRIDE];  // [m][k] bf16
  __shared__ __align__(16) unsigned short Bs[256 * BS_STRIDE];  // [n][k] bf16

  const int tid  = threadIdx.x;
  const int m0   = blockIdx.x * 64;
  const int w    = tid >> 5, lane = tid & 31;
  const int wrow = w & 3,  wcol = w >> 2;       // 4x2 wave grid
  const int half = lane >> 4, l16 = lane & 15;
  const bool fullM = (m0 + 64 <= Mdim);

  v8f c[8];
  v8f zv = {0.f, 0.f, 0.f, 0.f, 0.f, 0.f, 0.f, 0.f};
#pragma unroll
  for (int t = 0; t < 8; ++t) c[t] = zv;

  const int ksteps = (Kr + 31) >> 5;
  for (int ks = 0; ks < ksteps; ++ks) {
    const int k0 = ks << 5;
    const bool fullK = (k0 + 32 <= Kr);

    if (fullM && fullK) {
      // ======== fast path: branch-free staging, packed bf16x2 LDS stores ====
      if (AT) {
        // source S[k][m]; thread: m = tid&63, k pairs strided by 8
#pragma unroll
        for (int i = 0; i < 4; ++i) {
          const int m = tid & 63;
          const int k = ((tid >> 6) + i * 4) * 2;      // even, 0..30
          const float v0 = A[(size_t)(k0 + k)     * lda + (m0 + m)];
          const float v1 = A[(size_t)(k0 + k + 1) * lda + (m0 + m)];
          *(unsigned*)&As[m * AS_STRIDE + k] = pack2bf(v0, v1);
        }
      } else {
        // A row m, contiguous k: b64 global loads
#pragma unroll
        for (int i = 0; i < 4; ++i) {
          const int m = (tid >> 4) + i * 16;           // 0..63
          const int k = (tid & 15) * 2;                // even, 0..30
          const float2 v = *(const float2*)&A[(size_t)(m0 + m) * lda + (k0 + k)];
          *(unsigned*)&As[m * AS_STRIDE + k] = pack2bf(v.x, v.y);
        }
      }
      // B tile (32 k x 256 n) -> Bs[n][k]; thread owns column n = tid
      {
        const float* bp = B + (size_t)k0 * 256 + tid;
        unsigned* op = (unsigned*)&Bs[tid * BS_STRIDE];
#pragma unroll
        for (int j = 0; j < 16; ++j) {
          const float v0 = bp[(size_t)(2 * j)     * 256];
          const float v1 = bp[(size_t)(2 * j + 1) * 256];
          op[j] = pack2bf(v0, v1);
        }
      }
      // prefetch next k-step tile lines toward the WGP (global_prefetch_b8)
      if (ks + 1 < ksteps) {
        const int nk0 = k0 + 32;
        if (AT) {
          if (nk0 + ((tid >> 6) * 2) < Kr)
            __builtin_prefetch(&A[(size_t)(nk0 + (tid >> 6) * 2) * lda + m0 + (tid & 63)], 0, 1);
        } else {
          __builtin_prefetch(&A[(size_t)(m0 + (tid >> 4)) * lda + nk0 + (tid & 15) * 2], 0, 1);
        }
        if (nk0 < Kr)
          __builtin_prefetch(&B[(size_t)nk0 * 256 + tid], 0, 1);
      }
    } else {
      // ======== slow path: per-element guarded, zero-filled ================
      if (AT) {
#pragma unroll 1
        for (int i = 0; i < 8; ++i) {
          int e = i * 256 + tid;
          int k = e >> 6, m = e & 63;
          int gk = k0 + k, gm = m0 + m;
          float v = (gk < Kr && gm < Mdim) ? A[(size_t)gk * lda + gm] : 0.f;
          As[m * AS_STRIDE + k] = f2bf_bits(v);
        }
      } else {
#pragma unroll 1
        for (int i = 0; i < 8; ++i) {
          int e = i * 256 + tid;
          int k = e & 31, m = e >> 5;
          int gk = k0 + k, gm = m0 + m;
          float v = (gk < Kr && gm < Mdim) ? A[(size_t)gm * lda + gk] : 0.f;
          As[m * AS_STRIDE + k] = f2bf_bits(v);
        }
      }
#pragma unroll 1
      for (int i = 0; i < 32; ++i) {
        int gk = k0 + i;
        float v = (gk < Kr) ? B[(size_t)gk * 256 + tid] : 0.f;
        Bs[tid * BS_STRIDE + i] = f2bf_bits(v);
      }
    }
    __syncthreads();

    // A fragment (16x32 bf16): lane<16 -> K {0..7,16..23}, lane>=16 -> +8
    const int am = wrow * 16 + l16;
    const unsigned short* ap = &As[am * AS_STRIDE + half * 8];
    v16bf afrag = load_frag16(ap, ap + 16);

    // B fragments pipelined one tile ahead of the WMMA chain
    const int bbase = wcol * 128 + l16;
    const unsigned short* bp0 = &Bs[bbase * BS_STRIDE + half * 8];
    v16bf bfrag = load_frag16(bp0, bp0 + 16);
#pragma unroll
    for (int t = 0; t < 8; ++t) {
      v16bf bnext;
      if (t < 7) {
        const unsigned short* bp = &Bs[(bbase + (t + 1) * 16) * BS_STRIDE + half * 8];
        bnext = load_frag16(bp, bp + 16);
      }
      c[t] = __builtin_amdgcn_wmma_f32_16x16x32_bf16(
          false, afrag, false, bfrag, (short)0, c[t], false, false);
      if (t < 7) bfrag = bnext;
    }
    __syncthreads();
  }

  // epilogue: D layout VGPR v, lanes<16 -> M=v, lanes>=16 -> M=v+8, N=lane%16
#pragma unroll
  for (int t = 0; t < 8; ++t) {
    const int col = wcol * 128 + t * 16 + l16;
#pragma unroll
    for (int v = 0; v < 8; ++v) {
      const int row = m0 + wrow * 16 + v + 8 * half;
      if (row < Mdim) {
        float acc = c[t][v] * alpha;
        if (rowScale) acc *= rowScale[row];
        if (Cin)  acc += beta * Cin[(size_t)row * 256 + col];
        if (bias) acc += bias[col];
        C[(size_t)row * 256 + col] = acc;
      }
    }
  }
}

// ---------------------------------------------------------------------------
// Elementwise / reduction helper kernels
// ---------------------------------------------------------------------------
__global__ void zero_kernel(float* p, size_t n) {
  size_t i = (size_t)blockIdx.x * 256 + threadIdx.x;
  if (i < n) p[i] = 0.f;
}

__global__ void gather_kernel(const int* __restrict__ type_f,
                              const int* __restrict__ len_f,
                              const int* __restrict__ lane_f,
                              const float* __restrict__ node_tab,
                              const float* __restrict__ type_tab,
                              const float* __restrict__ len_tab,
                              const float* __restrict__ lane_tab,
                              float* __restrict__ raw, int N) {
  size_t idx = (size_t)blockIdx.x * 256 + threadIdx.x;
  if (idx >= (size_t)N * 256) return;
  int n = (int)(idx >> 8), c = (int)(idx & 255);
  float v;
  if      (c < 32)  v = lane_tab[lane_f[n] * 32 + c];
  else if (c < 64)  v = type_tab[type_f[n] * 32 + (c - 32)];
  else if (c < 96)  v = len_tab [len_f [n] * 32 + (c - 64)];
  else              v = node_tab[(size_t)n * 160 + (c - 96)];  // node_feature = arange
  raw[idx] = v;
}

__global__ void colsum_scale_kernel(const float* __restrict__ A, int rows, int cols,
                                    float* __restrict__ scale) {
  int k = blockIdx.x * 256 + threadIdx.x;
  if (k >= cols) return;
  float s = 0.f;
  for (int n = 0; n < rows; ++n) s += A[(size_t)n * cols + k];
  scale[k] = 1.f / (fmaxf(s - 1.f, 0.f) + 1.f);
}

__global__ void rowdiv_kernel(const float* __restrict__ fa0,
                              const float* __restrict__ scale_f,
                              float* __restrict__ fdiv, int K, int M) {
  int k = blockIdx.x * 256 + threadIdx.x;
  if (k >= K) return;
  const float* r = fa0 + (size_t)k * M;
  float s = 0.f;
  for (int m = 0; m < M; ++m) s += r[m] * scale_f[m];
  fdiv[k] = 1.f / (fmaxf(s - 1.f, 0.f) + 1.f);
}

// sadj = relu(struct_adj - 1e4*I) + I, then _gcn adds another I -> +2 on diag
__global__ void sadj_build_kernel(const float* __restrict__ adj, float* __restrict__ A, int n) {
  size_t idx = (size_t)blockIdx.x * 256 + threadIdx.x;
  if (idx >= (size_t)n * n) return;
  int i = (int)(idx / n), j = (int)(idx % n);
  float v = adj[idx] - (i == j ? 1e4f : 0.f);
  A[idx] = fmaxf(v, 0.f) + (i == j ? 2.f : 0.f);
}

// fnc_adj = sigmoid(fe fe^T) + I, then _gcn adds I -> +2 on diag
__global__ void fnc_adj_build_kernel(const float* __restrict__ fe, float* __restrict__ A, int M) {
  size_t idx = (size_t)blockIdx.x * 256 + threadIdx.x;
  if (idx >= (size_t)M * M) return;
  int i = (int)(idx / M), j = (int)(idx % M);
  const float* ri = fe + (size_t)i * 256;
  const float* rj = fe + (size_t)j * 256;
  float s = 0.f;
  for (int c = 0; c < 256; ++c) s += ri[c] * rj[c];
  A[idx] = 1.f / (1.f + expf(-s)) + (i == j ? 2.f : 0.f);
}

__global__ void deg_kernel(const float* __restrict__ A, float* __restrict__ d, int n) {
  int i = blockIdx.x * 256 + threadIdx.x;
  if (i >= n) return;
  const float* r = A + (size_t)i * n;
  float s = 0.f;
  for (int j = 0; j < n; ++j) s += fabsf(r[j]);
  d[i] = rsqrtf(s);
}

__global__ void adjnorm_kernel(float* __restrict__ A, const float* __restrict__ d, int n) {
  size_t idx = (size_t)blockIdx.x * 256 + threadIdx.x;
  if (idx >= (size_t)n * n) return;
  int i = (int)(idx / n), j = (int)(idx % n);
  A[idx] *= d[i] * d[j];
}

// s1[n] = h[n,:] . a[0:256], s2[n] = h[n,:] . a[256:512]  (one wave per node)
__global__ __launch_bounds__(256)
void rowdots_kernel(const float* __restrict__ h, const float* __restrict__ a,
                    float* __restrict__ s1, float* __restrict__ s2, int N) {
  int wv = (int)(((size_t)blockIdx.x * 256 + threadIdx.x) >> 5);
  int lane = threadIdx.x & 31;
  if (wv >= N) return;
  const float* r = h + (size_t)wv * 256;
  float d1 = 0.f, d2 = 0.f;
  for (int i = lane; i < 256; i += 32) {
    float x = r[i];
    d1 += x * a[i];
    d2 += x * a[256 + i];
  }
  for (int off = 16; off; off >>= 1) {
    d1 += __shfl_xor(d1, off, 32);
    d2 += __shfl_xor(d2, off, 32);
  }
  if (lane == 0) { s1[wv] = d1; s2[wv] = d2; }
}

__global__ __launch_bounds__(256)
void edge_max_kernel(const int* __restrict__ src, const int* __restrict__ dst,
                     const float* __restrict__ s1, const float* __restrict__ s2,
                     int E, unsigned* __restrict__ gmax) {
  int e = blockIdx.x * 256 + threadIdx.x;
  float v = -3.4e38f;
  if (e < E) {
    float x = s1[src[e]] + s2[dst[e]];
    v = x > 0.f ? x : 0.2f * x;          // leaky relu
  }
  for (int off = 16; off; off >>= 1) v = fmaxf(v, __shfl_xor(v, off, 32));
  if ((threadIdx.x & 31) == 0) atomicMax(gmax, enc_f32(v));
}

// one wave per edge: ev = exp(lrelu(s1[src]+s2[dst]) - max);
// rowsum[src] += ev; hp[src,:] += ev * h[dst,:]
__global__ __launch_bounds__(256)
void edge_scatter_kernel(const int* __restrict__ src, const int* __restrict__ dst,
                         const float* __restrict__ s1, const float* __restrict__ s2,
                         const float* __restrict__ h, float* __restrict__ hp,
                         float* __restrict__ rowsum, const unsigned* __restrict__ gmax,
                         int E) {
  int wv = (int)(((size_t)blockIdx.x * 256 + threadIdx.x) >> 5);
  int lane = threadIdx.x & 31;
  if (wv >= E) return;
  int s = src[wv], d = dst[wv];
  float mx = dec_f32(*gmax);
  float x = s1[s] + s2[d];
  x = x > 0.f ? x : 0.2f * x;
  float ev = expf(x - mx);
  if (lane == 0) atomicAdd(&rowsum[s], ev);
  const float* hd = h + (size_t)d * 256;
  float* o = hp + (size_t)s * 256;
#pragma unroll
  for (int i = 0; i < 8; ++i) {
    int col = lane + i * 32;
    atomicAdd(&o[col], ev * hd[col]);
  }
}

__global__ void out_elu_kernel(const float* __restrict__ hp, const float* __restrict__ rowsum,
                               float* __restrict__ out, int N) {
  size_t idx = (size_t)blockIdx.x * 256 + threadIdx.x;
  if (idx >= (size_t)N * 256) return;
  int n = (int)(idx >> 8);
  float v = hp[idx] / (rowsum[n] + 1e-15f);
  out[idx] = v > 0.f ? v : expm1f(v);
}

// ---------------------------------------------------------------------------
static inline int cdiv(long long a, long long b) { return (int)((a + b - 1) / b); }

extern "C" void kernel_launch(void* const* d_in, const int* in_sizes, int n_in,
                              void* d_out, int out_size, void* d_ws, size_t ws_size,
                              hipStream_t stream) {
  (void)n_in; (void)out_size; (void)ws_size;
  const int*   type_f     = (const int*)d_in[1];
  const int*   len_f      = (const int*)d_in[2];
  const int*   lane_f     = (const int*)d_in[3];
  const int*   eidx       = (const int*)d_in[4];
  const float* struct_adj = (const float*)d_in[5];
  const float* sa0        = (const float*)d_in[6];
  const float* fa0        = (const float*)d_in[7];
  const float* node_tab   = (const float*)d_in[8];
  const float* type_tab   = (const float*)d_in[9];
  const float* len_tab    = (const float*)d_in[10];
  const float* lane_tab   = (const float*)d_in[11];

  const int N = in_sizes[0];
  const int E = in_sizes[4] / 2;
  const int K = in_sizes[6] / N;
  const int M = in_sizes[7] / K;
  const int H = 256;

  // workspace carve (≈165 MB)
  char* wp = (char*)d_ws;
  auto take = [&](size_t bytes) -> char* {
    char* r = wp; wp += (bytes + 255) & ~(size_t)255; return r;
  };
  float* P     = (float*)take((size_t)N * H * 4);
  float* Q     = (float*)take((size_t)N * H * 4);
  float* R     = (float*)take((size_t)N * H * 4);
  float* semb  = (float*)take((size_t)1024 * H * 4);
  float* femb  = (float*)take((size_t)512 * H * 4);
  float* femb2 = (float*)take((size_t)512 * H * 4);
  float* zf    = (float*)take((size_t)512 * H * 4);
  float* zs    = (float*)take((size_t)1024 * H * 4);
  float* AnF   = (float*)take((size_t)M * M * 4);
  float* AnS   = (float*)take((size_t)K * K * 4);
  float* scale_s = (float*)take(4096);
  float* scale_f = (float*)take(4096);
  float* fdiv    = (float*)take(4096);
  float* dvec    = (float*)take(4096);
  float* s1      = (float*)take((size_t)N * 4);
  float* s2      = (float*)take((size_t)N * 4);
  float* rowsum  = (float*)take((size_t)N * 4);
  unsigned* gmax = (unsigned*)take(256);

  const int* esrc = eidx;
  const int* edst = eidx + E;

  // ---- layer-invariant precompute ----
  gather_kernel<<<cdiv((long long)N * H, 256), 256, 0, stream>>>(
      type_f, len_f, lane_f, node_tab, type_tab, len_tab, lane_tab, P, N);
  colsum_scale_kernel<<<cdiv(K, 256), 256, 0, stream>>>(sa0, N, K, scale_s);
  colsum_scale_kernel<<<cdiv(M, 256), 256, 0, stream>>>(fa0, K, M, scale_f);
  rowdiv_kernel<<<cdiv(K, 256), 256, 0, stream>>>(fa0, scale_f, fdiv, K, M);
  sadj_build_kernel<<<cdiv((long long)K * K, 256), 256, 0, stream>>>(struct_adj, AnS, K);
  deg_kernel<<<cdiv(K, 256), 256, 0, stream>>>(AnS, dvec, K);
  adjnorm_kernel<<<cdiv((long long)K * K, 256), 256, 0, stream>>>(AnS, dvec, K);

  auto layer = [&](const float* raw, float* rf, float* h, float* hp, float* out,
                   const float* fW, const float* fb, const float* sW, const float* sb,
                   const float* gW, const float* ga) {
    const int gK = cdiv(K, 64), gM = cdiv(M, 64), gN = cdiv(N, 64);
    // struct_emb = scale_s .* (sa0^T @ raw)
    gemm256<true ><<<gK, 256, 0, stream>>>(sa0, K, raw, semb, K, N,
                                           1.f, scale_s, 0.f, nullptr, nullptr);
    // fnc_emb = scale_f .* (fa0^T @ struct_emb)
    gemm256<true ><<<gM, 256, 0, stream>>>(fa0, M, semb, femb, M, K,
                                           1.f, scale_f, 0.f, nullptr, nullptr);
    // fnc GCN: An = norm(sigmoid(fe fe^T)+2I); femb2 = An @ (femb @ fW) + fb
    fnc_adj_build_kernel<<<cdiv((long long)M * M, 256), 256, 0, stream>>>(femb, AnF, M);
    deg_kernel<<<cdiv(M, 256), 256, 0, stream>>>(AnF, dvec, M);
    adjnorm_kernel<<<cdiv((long long)M * M, 256), 256, 0, stream>>>(AnF, dvec, M);
    gemm256<false><<<gM, 256, 0, stream>>>(femb, 256, fW, zf, M, 256,
                                           1.f, nullptr, 0.f, nullptr, nullptr);
    gemm256<false><<<gM, 256, 0, stream>>>(AnF, M, zf, femb2, M, M,
                                           1.f, nullptr, 0.f, nullptr, fb);
    // struct_emb += 0.15 * fdiv .* (fa0 @ femb2)
    gemm256<false><<<gK, 256, 0, stream>>>(fa0, M, femb2, semb, K, M,
                                           0.15f, fdiv, 1.f, semb, nullptr);
    // struct GCN: semb = AnS @ (semb @ sW) + sb
    gemm256<false><<<gK, 256, 0, stream>>>(semb, 256, sW, zs, K, 256,
                                           1.f, nullptr, 0.f, nullptr, nullptr);
    gemm256<false><<<gK, 256, 0, stream>>>(AnS, K, zs, semb, K, K,
                                           1.f, nullptr, 0.f, nullptr, sb);
    // rf = raw + 0.5 * (sa0 @ semb)
    gemm256<false><<<gN, 256, 0, stream>>>(sa0, K, semb, rf, N, K,
                                           0.5f, nullptr, 1.f, raw, nullptr);
    // h = rf @ gW
    gemm256<false><<<gN, 256, 0, stream>>>(rf, 256, gW, h, N, 256,
                                           1.f, nullptr, 0.f, nullptr, nullptr);
    // sparse GAT
    rowdots_kernel<<<cdiv((long long)N * 32, 256), 256, 0, stream>>>(h, ga, s1, s2, N);
    zero_kernel<<<cdiv(N, 256), 256, 0, stream>>>(rowsum, (size_t)N);
    zero_kernel<<<cdiv((long long)N * H, 256), 256, 0, stream>>>(hp, (size_t)N * H);
    zero_kernel<<<1, 256, 0, stream>>>((float*)gmax, 1);
    edge_max_kernel<<<cdiv(E, 256), 256, 0, stream>>>(esrc, edst, s1, s2, E, gmax);
    edge_scatter_kernel<<<cdiv((long long)E * 32, 256), 256, 0, stream>>>(
        esrc, edst, s1, s2, h, hp, rowsum, gmax, E);
    out_elu_kernel<<<cdiv((long long)N * H, 256), 256, 0, stream>>>(hp, rowsum, out, N);
  };

  const float* fW1 = (const float*)d_in[12]; const float* fb1 = (const float*)d_in[13];
  const float* sW1 = (const float*)d_in[14]; const float* sb1 = (const float*)d_in[15];
  const float* gW1 = (const float*)d_in[16]; const float* ga1 = (const float*)d_in[17];
  const float* fW2 = (const float*)d_in[18]; const float* fb2 = (const float*)d_in[19];
  const float* sW2 = (const float*)d_in[20]; const float* sb2 = (const float*)d_in[21];
  const float* gW2 = (const float*)d_in[22]; const float* ga2 = (const float*)d_in[23];

  // Buffer ping-pong: layer1 raw=P, rf=Q, h=P (raw dead), hp=R, out=Q.
  layer(P, Q, P, R, Q, fW1, fb1, sW1, sb1, gW1, ga1);
  // layer2 raw=Q, rf=P, h=R, hp=Q (raw dead), out=d_out.
  layer(Q, P, R, Q, (float*)d_out, fW2, fb2, sW2, sb2, gW2, ga2);
}
